// Llama4TextMoELayer_44710609551569
// MI455X (gfx1250) — compile-verified, matrix operand order checked by reading, not run on hardware
//
#include <hip/hip_runtime.h>
#include <math.h>

// ---------------------------------------------------------------- constants
#define SEQ 1024
#define BS  2
#define T   2048          // SEQ*BS tokens
#define D   2048
#define F   2048
#define E   8
#define TP  3072          // T + E*128 padded rows (worst case)
#define MT_PAD 24         // TP / 128 row tiles

#define BM 128
#define BN 128
#define BK 32
#define LDSP 40           // LDS row pitch in elements (80B, 16B aligned)

typedef __bf16 v16bf __attribute__((ext_vector_type(16)));
typedef __bf16 v8bf  __attribute__((ext_vector_type(8)));
typedef float  v8f   __attribute__((ext_vector_type(8)));

// ---------------------------------------------------------------- helpers
// Native f32<->bf16 converts (lowers to v_cvt_*bf16* on gfx1250, RTNE)
__device__ __forceinline__ unsigned short f2bf_u(float f) {
    __bf16 h = (__bf16)f;
    return __builtin_bit_cast(unsigned short, h);
}
__device__ __forceinline__ float bf2f(unsigned short h) {
    return (float)__builtin_bit_cast(__bf16, h);
}
// Build a 16x(K=32) bf16 fragment from k-contiguous LDS (ISA 7.12.2 layout):
// lane holds K = kb..kb+7 and kb+16..kb+23 (kb = 0 for lanes 0-15, 8 for 16-31)
__device__ __forceinline__ v16bf load_frag16(const unsigned short* p) {
    v8bf lo = *reinterpret_cast<const v8bf*>(p);
    v8bf hi = *reinterpret_cast<const v8bf*>(p + 16);
    return __builtin_shufflevector(lo, hi, 0,1,2,3,4,5,6,7,8,9,10,11,12,13,14,15);
}
// CDNA5 async copy global->LDS, 16B per lane, tracked by ASYNCcnt.
// LDS operand VGPR holds the wave-relative LDS byte address (= low 32 bits of
// the generic shared-aperture pointer).
__device__ __forceinline__ void async_g2l_b128(void* lds_ptr, const void* gptr) {
    unsigned lds_off = (unsigned)(unsigned long long)lds_ptr;
    asm volatile("global_load_async_to_lds_b128 %0, %1, off"
                 :: "v"(lds_off), "v"(gptr)
                 : "memory");
}
__device__ __forceinline__ void wait_asynccnt0() {
    asm volatile("s_wait_asynccnt 0x0" ::: "memory");
}

// ---------------------------------------------------------------- router
__global__ __launch_bounds__(256)
void router_kernel(const float* __restrict__ x, const float* __restrict__ rw,
                   int* __restrict__ eidx, float* __restrict__ score,
                   unsigned short* __restrict__ xbf, int* __restrict__ counts) {
    int token = blockIdx.x * 8 + (threadIdx.x >> 5);
    int lane  = threadIdx.x & 31;
    float acc[E];
#pragma unroll
    for (int e = 0; e < E; ++e) acc[e] = 0.0f;
    const float* xr = x + (long long)token * D;
    for (int d = lane; d < D; d += 32) {
        float xv = xr[d];
        xbf[(long long)token * D + d] = f2bf_u(xv);   // bf16 copy for shared path
#pragma unroll
        for (int e = 0; e < E; ++e) acc[e] += xv * rw[d * E + e];
    }
#pragma unroll
    for (int off = 16; off > 0; off >>= 1)
#pragma unroll
        for (int e = 0; e < E; ++e) acc[e] += __shfl_down(acc[e], off, 32);
    if (lane == 0) {
        int best = 0; float bv = acc[0];
#pragma unroll
        for (int e = 1; e < E; ++e) if (acc[e] > bv) { bv = acc[e]; best = e; }
        eidx[token]  = best;
        score[token] = 1.0f / (1.0f + expf(-bv));
        atomicAdd(&counts[best], 1);
    }
}

__global__ void init_meta(int* counts, int* cursor) {
    int i = threadIdx.x;
    if (i < E) { counts[i] = 0; cursor[i] = 0; }
}

// one thread: padded per-expert segment offsets + tile->expert map
__global__ void plan_kernel(const int* __restrict__ counts,
                            int* __restrict__ pad_off, int* __restrict__ tile_expert) {
    if (threadIdx.x == 0 && blockIdx.x == 0) {
        int run = 0;
        for (int e = 0; e < E; ++e) {
            pad_off[e] = run;
            int tiles = (counts[e] + BM - 1) >> 7;
            for (int i = 0; i < tiles; ++i) tile_expert[(run >> 7) + i] = e;
            run += tiles << 7;
        }
        for (int t = run >> 7; t < MT_PAD; ++t) tile_expert[t] = -1;
    }
}

// gather tokens into expert-sorted, padded rows; apply sigmoid scale; f32->bf16
__global__ __launch_bounds__(256)
void scatter_kernel(const float* __restrict__ x, const float* __restrict__ score,
                    const int* __restrict__ eidx, const int* __restrict__ pad_off,
                    int* __restrict__ cursor, int* __restrict__ pos,
                    unsigned short* __restrict__ xs_perm) {
    int t = blockIdx.x;
    __shared__ int sp;
    if (threadIdx.x == 0) {
        int e = eidx[t];
        sp = pad_off[e] + atomicAdd(&cursor[e], 1);
        pos[t] = sp;
    }
    __syncthreads();
    int p = sp;
    float s = score[t];
    const float* xr = x + (long long)t * D;
    unsigned short* dst = xs_perm + (long long)p * D;
    for (int d = threadIdx.x; d < D; d += 256) dst[d] = f2bf_u(xr[d] * s);
}

// ---------------------------------------------------------------- WMMA GEMM
// C[M,N] = A[M,K](bf16) @ B[K,N](f32 weights, bf16-converted during staging).
// Double-buffered LDS; A staged via GLOBAL_LOAD_ASYNC_TO_LDS_B128 (ASYNCcnt),
// B loaded to registers for the next tile while WMMAs run on the current one.
// tile_expert != nullptr : per-row-tile expert selection (B += e*bstride)
// addend/pos != nullptr  : C[r,:] += addend[pos[r]*ldc + :]
__global__ __launch_bounds__(256)
void gemm_bf16_kernel(const unsigned short* __restrict__ A, int lda,
                      const float* __restrict__ Bbase, long long bstride,
                      const int* __restrict__ tile_expert,
                      int K, int ldb,
                      void* __restrict__ Cout, int ldc, int c_bf16,
                      const float* __restrict__ addend, const int* __restrict__ pos) {
    int mtile = blockIdx.y, ntile = blockIdx.x;
    const float* Bw = Bbase;
    if (tile_expert) {
        int e = tile_expert[mtile];
        if (e < 0) return;                       // inactive padded tile
        Bw = Bbase + (long long)e * bstride;
    }
    __shared__ unsigned short Alds[2][BM * LDSP];
    __shared__ unsigned short Blds[2][BN * LDSP];   // column-major: [n][k]

    int tid  = threadIdx.x;
    int lane = tid & 31, wave = tid >> 5;
    int wm = wave & 3, wn = wave >> 2;           // 4x2 waves -> 32x64 per wave
    int half = lane >> 4, r16 = lane & 15;
    int kb = half * 8;

    int m0 = mtile * BM, n0 = ntile * BN;
    // A staging: 2 threads/row, 16 bf16 (32B = 2x b128 async) each
    int arow = tid >> 1, akoff = (tid & 1) * 16;
    // B staging: thread owns k-pair (2 rows) x 8 columns -> 8 packed b32 stores
    int bkp = tid >> 4, bns = (tid & 15) * 8;

    const unsigned short* agp = A + (long long)(m0 + arow) * lda + akoff;
    const float* bgp = Bw + (long long)(2 * bkp) * ldb + n0 + bns;

    v8f acc[2][4];
#pragma unroll
    for (int i = 0; i < 2; ++i)
#pragma unroll
        for (int j = 0; j < 4; ++j) acc[i][j] = (v8f)(0.0f);

    float b0[8], b1[8];

    auto stageA = [&](int buf, int k0) {
        async_g2l_b128(&Alds[buf][arow * LDSP + akoff],     agp + k0);
        async_g2l_b128(&Alds[buf][arow * LDSP + akoff + 8], agp + k0 + 8);
    };
    auto loadB = [&](int k0) {
        const float* r0 = bgp + (long long)k0 * ldb;
        const float* r1 = r0 + ldb;
#pragma unroll
        for (int i = 0; i < 8; ++i) { b0[i] = r0[i]; b1[i] = r1[i]; }
        __builtin_prefetch(r0 + (long long)2 * BK * ldb, 0, 0);  // 2 tiles ahead
    };
    auto storeB = [&](int buf) {
#pragma unroll
        for (int i = 0; i < 8; ++i) {
            unsigned v = (unsigned)f2bf_u(b0[i]) | ((unsigned)f2bf_u(b1[i]) << 16);
            *reinterpret_cast<unsigned*>(&Blds[buf][(bns + i) * LDSP + 2 * bkp]) = v;
        }
    };

    // prologue: stage K-tile 0
    stageA(0, 0);
    loadB(0);
    storeB(0);
    wait_asynccnt0();
    __syncthreads();

    int nk = K / BK;
    for (int kt = 0; kt < nk; ++kt) {
        int cur = kt & 1, nxt = cur ^ 1;
        if (kt + 1 < nk) {               // kick off next tile (async A, B->regs)
            stageA(nxt, (kt + 1) * BK);
            loadB((kt + 1) * BK);
        }

        v16bf afr[2], bfr[4];
#pragma unroll
        for (int i = 0; i < 2; ++i)
            afr[i] = load_frag16(&Alds[cur][(wm * 32 + i * 16 + r16) * LDSP + kb]);
#pragma unroll
        for (int j = 0; j < 4; ++j)
            bfr[j] = load_frag16(&Blds[cur][(wn * 64 + j * 16 + r16) * LDSP + kb]);

#pragma unroll
        for (int i = 0; i < 2; ++i)
#pragma unroll
            for (int j = 0; j < 4; ++j)
                acc[i][j] = __builtin_amdgcn_wmma_f32_16x16x32_bf16(
                    false, afr[i], false, bfr[j], (short)0, acc[i][j], false, false);

        if (kt + 1 < nk) {
            storeB(nxt);                 // convert+pack B into the other buffer
            wait_asynccnt0();            // A async copies of nxt complete
        }
        __syncthreads();
    }

    // --- epilogue: ISA C layout (VGPR r: M = r + half*8, N = lane%16) ---
#pragma unroll
    for (int i = 0; i < 2; ++i)
#pragma unroll
        for (int j = 0; j < 4; ++j) {
            int ncol = n0 + wn * 64 + j * 16 + r16;
#pragma unroll
            for (int r = 0; r < 8; ++r) {
                int grow = m0 + wm * 32 + i * 16 + half * 8 + r;
                float v = acc[i][j][r];
                if (addend) v += addend[(long long)pos[grow] * ldc + ncol];
                if (c_bf16)
                    ((unsigned short*)Cout)[(long long)grow * ldc + ncol] = f2bf_u(v);
                else
                    ((float*)Cout)[(long long)grow * ldc + ncol] = v;
            }
        }
}

// ---------------------------------------------------------------- silu * up
__global__ __launch_bounds__(256)
void silu_mul_kernel(const unsigned short* __restrict__ G,
                     const unsigned short* __restrict__ U,
                     unsigned short* __restrict__ H, long long n) {
    long long i = (long long)blockIdx.x * blockDim.x + threadIdx.x;
    long long stride = (long long)gridDim.x * blockDim.x;
    for (; i < n; i += stride) {
        float g = bf2f(G[i]), u = bf2f(U[i]);
        float s = g / (1.0f + expf(-g));
        H[i] = f2bf_u(s * u);
    }
}

// ---------------------------------------------------------------- launch
extern "C" void kernel_launch(void* const* d_in, const int* in_sizes, int n_in,
                              void* d_out, int out_size, void* d_ws, size_t ws_size,
                              hipStream_t stream) {
    const float* x  = (const float*)d_in[0];   // [T, D]
    const float* rw = (const float*)d_in[1];   // [D, E]
    const float* gw = (const float*)d_in[2];   // [E, D, F]
    const float* uw = (const float*)d_in[3];   // [E, D, F]
    const float* dw = (const float*)d_in[4];   // [E, F, D]
    const float* sg = (const float*)d_in[5];   // [D, F]
    const float* su = (const float*)d_in[6];   // [D, F]
    const float* sd = (const float*)d_in[7];   // [F, D]
    float* out = (float*)d_out;                // [T, D] f32

    char* w = (char*)d_ws;
    int*   counts  = (int*)(w);
    int*   cursor  = (int*)(w + 64);
    int*   pad_off = (int*)(w + 128);
    int*   tilee   = (int*)(w + 192);
    int*   eidx    = (int*)(w + 1024);
    float* score   = (float*)(w + 1024 + 8192);
    int*   pos     = (int*)(w + 1024 + 16384);

    size_t base = 65536;
    unsigned short* xbf = (unsigned short*)(w + base); base += (size_t)T  * D * 2;
    unsigned short* xsp = (unsigned short*)(w + base); base += (size_t)TP * D * 2;
    unsigned short* Gp  = (unsigned short*)(w + base); base += (size_t)TP * F * 2;
    unsigned short* Up  = (unsigned short*)(w + base); base += (size_t)TP * F * 2;
    unsigned short* Hp  = (unsigned short*)(w + base); base += (size_t)TP * F * 2;
    unsigned short* Gs  = (unsigned short*)(w + base); base += (size_t)T  * F * 2;
    unsigned short* Us  = (unsigned short*)(w + base); base += (size_t)T  * F * 2;
    unsigned short* Hs  = (unsigned short*)(w + base); base += (size_t)T  * F * 2;
    float*          eout= (float*)(w + base);          base += (size_t)TP * D * 4;

    dim3 blk(256);
    dim3 gE(F / BN, MT_PAD);   // expert GEMMs over padded rows
    dim3 gS(F / BN, T / BM);   // shared GEMMs

    init_meta<<<1, 32, 0, stream>>>(counts, cursor);
    router_kernel<<<T / 8, 256, 0, stream>>>(x, rw, eidx, score, xbf, counts);
    plan_kernel<<<1, 1, 0, stream>>>(counts, pad_off, tilee);
    scatter_kernel<<<T, 256, 0, stream>>>(x, score, eidx, pad_off, cursor, pos, xsp);

    // routed expert: gate & up  (B selected per row-tile by expert id)
    gemm_bf16_kernel<<<gE, blk, 0, stream>>>(xsp, D, gw, (long long)D * F, tilee,
                                             D, F, Gp, F, 1, nullptr, nullptr);
    gemm_bf16_kernel<<<gE, blk, 0, stream>>>(xsp, D, uw, (long long)D * F, tilee,
                                             D, F, Up, F, 1, nullptr, nullptr);
    silu_mul_kernel<<<2048, 256, 0, stream>>>(Gp, Up, Hp, (long long)TP * F);

    // shared expert: gate & up
    gemm_bf16_kernel<<<gS, blk, 0, stream>>>(xbf, D, sg, 0, nullptr,
                                             D, F, Gs, F, 1, nullptr, nullptr);
    gemm_bf16_kernel<<<gS, blk, 0, stream>>>(xbf, D, su, 0, nullptr,
                                             D, F, Us, F, 1, nullptr, nullptr);
    silu_mul_kernel<<<2048, 256, 0, stream>>>(Gs, Us, Hs, (long long)T * F);

    // routed expert: down -> f32 expert output (permuted rows)
    gemm_bf16_kernel<<<gE, blk, 0, stream>>>(Hp, F, dw, (long long)F * D, tilee,
                                             F, D, eout, D, 0, nullptr, nullptr);
    // shared down + combine with routed output (gather via pos) -> d_out
    gemm_bf16_kernel<<<dim3(D / BN, T / BM), blk, 0, stream>>>(
        Hs, F, sd, 0, nullptr, F, D, out, D, 0, eout, pos);

    (void)in_sizes; (void)n_in; (void)out_size; (void)ws_size;
}
